// HybridGAT_SplineNet_40535901339690
// MI455X (gfx1250) — matrix operand order repaired and checked by reading, not compiled, not run on hardware
//
#include <hip/hip_runtime.h>
#include <math.h>

#define NN 50000
#define EE 400000
#define ETOT (EE + NN)
#define KSP 5
#define MT 5                    // M-tiles (16 rows each) per block: 80 rows, 50000 = 625*80

typedef __attribute__((ext_vector_type(16))) __bf16 v16bf;
typedef __attribute__((ext_vector_type(8)))  float  v8f;
typedef __attribute__((ext_vector_type(4)))  unsigned int u32x4;

union V16U { v16bf v; u32x4 q[2]; unsigned short u[16]; };

static inline int ceil_div(long long a, long long b) { return (int)((a + b - 1) / b); }

// ---------------- device helpers ----------------

__device__ __forceinline__ float bf2f(unsigned short h) {
    return __int_as_float((int)(((unsigned int)h) << 16));
}
__device__ __forceinline__ unsigned short f2bf(float f) {
    unsigned int u = (unsigned int)__float_as_int(f);
    unsigned int r = u + 0x7fffu + ((u >> 16) & 1u);   // round-to-nearest-even
    return (unsigned short)(r >> 16);
}
// float atomic max via monotone int mapping (valid for -inf init, no NaNs)
__device__ __forceinline__ void atomicMaxF(float* addr, float val) {
    if (val >= 0.0f) atomicMax((int*)addr, __float_as_int(val));
    else             atomicMin((unsigned int*)addr, (unsigned int)__float_as_int(val));
}
__device__ __forceinline__ float eluf(float v)   { return v > 0.0f ? v : __expf(v) - 1.0f; }
__device__ __forceinline__ float lrelu(float v)  { return v > 0.0f ? v : 0.2f * v; }

// async DMA of one dword global -> LDS (CDNA5, ASYNCcnt-tracked)
__device__ __forceinline__ void async_g2l_b32(unsigned int lds_addr, const unsigned int* gp) {
    asm volatile("global_load_async_to_lds_b32 %0, %1, off"
                 :: "v"(lds_addr), "v"((unsigned long long)gp) : "memory");
}
__device__ __forceinline__ void wait_async0() {
    asm volatile("s_wait_asynccnt 0" ::: "memory");
}

// ---------------- format conversion / packing ----------------

__global__ void cvt_pad(const float* __restrict__ src, unsigned short* __restrict__ dst,
                        int rows, int cols, int rows_pad, int cols_pad) {
    long long t = (long long)blockIdx.x * blockDim.x + threadIdx.x;
    long long tot = (long long)rows_pad * cols_pad;
    if (t >= tot) return;
    int r = (int)(t / cols_pad), c = (int)(t % cols_pad);
    float v = (r < rows && c < cols) ? src[(long long)r * cols + c] : 0.0f;
    dst[t] = f2bf(v);
}

// Ws1[25][256][64] -> B[256][1600] with B[i][k*64+o] = Ws1[k][i][o]
__global__ void repack_ws1(const float* __restrict__ src, unsigned short* __restrict__ dst) {
    long long t = (long long)blockIdx.x * blockDim.x + threadIdx.x;
    if (t >= 256LL * 1600LL) return;
    int i = (int)(t / 1600), rest = (int)(t % 1600);
    int k = rest / 64, o = rest % 64;
    dst[t] = f2bf(src[((long long)k * 256 + i) * 64 + o]);
}
// Ws2[25][512][1] -> B[512][32] with B[i][k] = Ws2[k][i], zero-pad k>=25
__global__ void repack_ws2(const float* __restrict__ src, unsigned short* __restrict__ dst) {
    int t = blockIdx.x * blockDim.x + threadIdx.x;
    if (t >= 512 * 32) return;
    int i = t / 32, k = t % 32;
    float v = (k < 25) ? src[(long long)k * 512 + i] : 0.0f;
    dst[t] = f2bf(v);
}

// pack row-major bf16 B[Kdim x Npad] into this file's WMMA B-fragment layout (contiguous k, same order as A):
// element e of lane l in block (kt,nt) <- B[kt*32 + (l>>4)*16 + e][nt*16 + (l&15)]
__global__ void pack_b_tiles(const unsigned short* __restrict__ Brm, unsigned short* __restrict__ Bt,
                             int Kdim, int Npad) {
    int Ntiles = Npad >> 4;
    long long tot = (long long)(Kdim >> 5) * Ntiles * 512;
    long long t = (long long)blockIdx.x * blockDim.x + threadIdx.x;
    if (t >= tot) return;
    int e    = (int)(t & 15);
    int lane = (int)((t >> 4) & 31);
    long long blk = t >> 9;
    int nt = (int)(blk % Ntiles);
    int kt = (int)(blk / Ntiles);
    int k = kt * 32 + (lane >> 4) * 16 + e;
    int n = nt * 16 + (lane & 15);
    Bt[t] = Brm[(size_t)k * Npad + n];
}

__global__ void fill_f32(float* __restrict__ p, long long n, float val) {
    long long t = (long long)blockIdx.x * blockDim.x + threadIdx.x;
    if (t < n) p[t] = val;
}
__global__ void fix_neginf(float* __restrict__ p, long long n) {
    long long t = (long long)blockIdx.x * blockDim.x + threadIdx.x;
    if (t < n && p[t] == -INFINITY) p[t] = 0.0f;
}

// ---------------- WMMA GEMM ----------------
// C[M x Npad] = A[M x Kdim]bf16 @ B (pre-packed tiles, see pack_b_tiles).
// Requires: M % (16*MT) == 0 (M = 50000 = 625*80), Kdim % 32 == 0, Npad % 16 == 0.
// Block = 256 thr = 8 waves; each wave owns MT 16x16 C tiles in one 16-col stripe.
// Double-buffered LDS: the k+1 A tile is DMA'd (global_load_async_to_lds_b32)
// while the k tile feeds 5 WMMAs; one barrier per k-step. A fragments are
// software-pipelined one deep so ds_loads fly in the shadow of the WMMAs.
template <bool OUT_BF16>
__global__ __launch_bounds__(256) void wmma_gemm(const unsigned short* __restrict__ A,
                                                 const unsigned short* __restrict__ Bt,
                                                 void* __restrict__ Cout,
                                                 int Npad, int Kdim) {
    __shared__ __align__(16) unsigned int As[2 * MT * 256];   // 2 x MT x (16 rows x 16 dwords)

    int tid   = threadIdx.x;
    int wave  = tid >> 5;
    int lane  = tid & 31;
    int row_blk = blockIdx.x * (16 * MT);
    int ntile = blockIdx.y * 8 + wave;
    int Ntiles = Npad >> 4;
    bool active = (ntile * 16 < Npad);          // wave-uniform: EXEC all-ones inside

    // A-fragment LDS quad index: row r occupies uint4 [4r..4r+3];
    // lanes 0-15 take quads {0,1}, lanes 16-31 take {2,3} (contiguous K-halves)
    int r4 = (lane & 15) * 4 + (lane >> 4) * 2;

    const unsigned short* Bp = Bt + ((size_t)ntile * 32 + lane) * 16;
    size_t bstep = (size_t)Ntiles * 512;        // shorts per k-step across all ntiles

    // staging: thread t covers dword (t&15) of row (t>>4) in each of the MT tiles
    int srow = tid >> 4, scol = tid & 15;
    unsigned int lds_w = (unsigned int)(unsigned long long)&As[tid];

    // prologue: stage k0 = 0 into parity buffer 0
#pragma unroll
    for (int m = 0; m < MT; ++m) {
        const unsigned int* gp =
            (const unsigned int*)(A + (size_t)(row_blk + m * 16 + srow) * Kdim) + scol;
        async_g2l_b32(lds_w + (unsigned)(m * 1024), gp);
    }
    wait_async0();
    __syncthreads();

    v8f acc[MT] = {};
    for (int k0 = 0; k0 < Kdim; k0 += 32) {
        int par = (k0 >> 5) & 1;
        if (k0 + 32 < Kdim) {                   // DMA next tile into other parity
            unsigned int dstb = lds_w + (unsigned)((par ^ 1) * (MT * 1024));
#pragma unroll
            for (int m = 0; m < MT; ++m) {
                const unsigned int* gp =
                    (const unsigned int*)(A + (size_t)(row_blk + m * 16 + srow) * Kdim + k0 + 32) + scol;
                async_g2l_b32(dstb + (unsigned)(m * 1024), gp);
            }
        }
        if (active) {
            V16U b;
            const u32x4* Bq = (const u32x4*)Bp;
            b.q[0] = Bq[0];
            b.q[1] = Bq[1];
            if (k0 + 32 < Kdim)
                __builtin_prefetch((const void*)(Bp + bstep), 0, 3);
            const u32x4* Ap = (const u32x4*)(As + par * (MT * 256));
            V16U a0, an;
            a0.q[0] = Ap[r4];
            a0.q[1] = Ap[r4 + 1];
#pragma unroll
            for (int m = 0; m < MT; ++m) {
                if (m + 1 < MT) {               // fetch next fragment under this WMMA
                    an.q[0] = Ap[(m + 1) * 64 + r4];
                    an.q[1] = Ap[(m + 1) * 64 + r4 + 1];
                }
                acc[m] = __builtin_amdgcn_wmma_f32_16x16x32_bf16(false, a0.v, false, b.v,
                                                                 (short)0, acc[m], false, false);
                a0 = an;
            }
        }
        Bp += bstep;
        wait_async0();                           // next tile landed in LDS
        __syncthreads();                         // publish to all waves / fence reuse
    }
    if (active) {
        int ccol = ntile * 16 + (lane & 15);
#pragma unroll
        for (int m = 0; m < MT; ++m) {
            int crow0 = row_blk + m * 16 + ((lane >> 4) * 8);
#pragma unroll
            for (int v = 0; v < 8; ++v) {
                size_t idx = (size_t)(crow0 + v) * Npad + ccol;
                if (OUT_BF16) ((unsigned short*)Cout)[idx] = f2bf(acc[m][v]);
                else          ((float*)Cout)[idx]          = acc[m][v];
            }
        }
    }
}

// ---------------- GAT kernels ----------------

__global__ void gat_node_scores(const float* __restrict__ xp,
                                const float* __restrict__ a_src,
                                const float* __restrict__ a_dst,
                                float* __restrict__ es, float* __restrict__ ed, int C) {
    int t = blockIdx.x * blockDim.x + threadIdx.x;
    if (t >= NN * 8) return;
    int n = t >> 3, h = t & 7;
    const float* row = xp + ((size_t)n * 8 + h) * C;
    float s1 = 0.f, s2 = 0.f;
    for (int c = 0; c < C; ++c) {
        float v = row[c];
        s1 += v * a_src[h * C + c];
        s2 += v * a_dst[h * C + c];
    }
    es[t] = s1; ed[t] = s2;
}

__global__ void gat_edge_logits(const int* __restrict__ ei,
                                const float* __restrict__ es, const float* __restrict__ ed,
                                float* __restrict__ eatt, float* __restrict__ mbuf) {
    int t = blockIdx.x * blockDim.x + threadIdx.x;
    if (t >= ETOT * 8) return;
    int e = t >> 3, h = t & 7;
    int s, d;
    if (e < EE) { s = ei[e]; d = ei[EE + e]; } else { s = d = e - EE; }
    float v = lrelu(es[s * 8 + h] + ed[d * 8 + h]);
    eatt[t] = v;
    atomicMaxF(&mbuf[d * 8 + h], v);
}

__global__ void gat_edge_expsum(const int* __restrict__ ei, const float* __restrict__ mbuf,
                                float* __restrict__ eatt, float* __restrict__ sbuf) {
    int t = blockIdx.x * blockDim.x + threadIdx.x;
    if (t >= ETOT * 8) return;
    int e = t >> 3, h = t & 7;
    int d = (e < EE) ? ei[EE + e] : (e - EE);
    float ex = __expf(eatt[t] - mbuf[d * 8 + h]);
    eatt[t] = ex;
    atomicAdd(&sbuf[d * 8 + h], ex);
}

// one block per edge; blockDim = H*C
__global__ void gat_scatter(const int* __restrict__ ei, const float* __restrict__ eatt,
                            const float* __restrict__ sbuf, const float* __restrict__ xp,
                            float* __restrict__ acc, int HC, int cshift) {
    int e = blockIdx.x;
    int f = threadIdx.x;
    int s, d;
    if (e < EE) { s = ei[e]; d = ei[EE + e]; } else { s = d = e - EE; }
    int h = f >> cshift;
    float alpha = eatt[(size_t)e * 8 + h] / fmaxf(sbuf[d * 8 + h], 1e-16f);
    atomicAdd(&acc[(size_t)d * HC + f], xp[(size_t)s * HC + f] * alpha);
}

__global__ void gat_finalize(const float* __restrict__ acc, const float* __restrict__ b,
                             unsigned short* __restrict__ hout, int HC) {
    long long t = (long long)blockIdx.x * blockDim.x + threadIdx.x;
    if (t >= (long long)NN * HC) return;
    int f = (int)(t % HC);
    hout[t] = f2bf(eluf(acc[t] + b[f]));
}

// ---------------- Spline kernels ----------------

__global__ void spline_basis_k(const float* __restrict__ eattr,
                               float* __restrict__ basis, int* __restrict__ bidx) {
    int e = blockIdx.x * blockDim.x + threadIdx.x;
    if (e >= EE) return;
    float p0 = eattr[2 * e], p1 = eattr[2 * e + 1];
    float v0 = p0 * (KSP - 1), v1 = p1 * (KSP - 1);
    float fl0 = floorf(v0), fl1 = floorf(v1);
    float fr0 = v0 - fl0, fr1 = v1 - fl1;
    int i0 = min(max((int)fl0, 0), KSP - 1), i1 = min(i0 + 1, KSP - 1);
    int j0 = min(max((int)fl1, 0), KSP - 1), j1 = min(j0 + 1, KSP - 1);
    float w0[2] = {1.0f - fr0, fr0};
    float w1[2] = {1.0f - fr1, fr1};
    int id0[2] = {i0, i1}, id1[2] = {j0, j1};
#pragma unroll
    for (int a = 0; a < 2; ++a)
#pragma unroll
        for (int b = 0; b < 2; ++b) {
            basis[(size_t)e * 4 + a * 2 + b] = w0[a] * w1[b];
            bidx [(size_t)e * 4 + a * 2 + b] = id0[a] + KSP * id1[b];
        }
}

__global__ void spline1_edge(const int* __restrict__ ei, const float* __restrict__ basis,
                             const int* __restrict__ bidx, const unsigned short* __restrict__ xk,
                             float* __restrict__ agg) {
    long long t = (long long)blockIdx.x * blockDim.x + threadIdx.x;
    if (t >= (long long)EE * 64) return;
    int e = (int)(t >> 6), o = (int)(t & 63);
    int s = ei[e], d = ei[EE + e];
    const float* bb = &basis[(size_t)e * 4];
    const int*   ii = &bidx[(size_t)e * 4];
    float m = 0.f;
#pragma unroll
    for (int j = 0; j < 4; ++j)
        m += bb[j] * bf2f(xk[(size_t)s * 1600 + ii[j] * 64 + o]);
    atomicMaxF(&agg[(size_t)d * 64 + o], m);
}

__global__ void spline1_finalize(const float* __restrict__ agg, const float* __restrict__ rootout,
                                 const float* __restrict__ bs, unsigned short* __restrict__ hout) {
    long long t = (long long)blockIdx.x * blockDim.x + threadIdx.x;
    if (t >= (long long)NN * 64) return;
    int o = (int)(t & 63);
    float m = agg[t];
    if (m == -INFINITY) m = 0.f;
    hout[t] = f2bf(eluf(m + rootout[t] + bs[o]));
}

__global__ void spline2_edge(const int* __restrict__ ei, const float* __restrict__ basis,
                             const int* __restrict__ bidx, const unsigned short* __restrict__ xk2,
                             float* __restrict__ agg2) {
    int e = blockIdx.x * blockDim.x + threadIdx.x;
    if (e >= EE) return;
    int s = ei[e], d = ei[EE + e];
    const float* bb = &basis[(size_t)e * 4];
    const int*   ii = &bidx[(size_t)e * 4];
    float m = 0.f;
#pragma unroll
    for (int j = 0; j < 4; ++j)
        m += bb[j] * bf2f(xk2[(size_t)s * 32 + ii[j]]);
    atomicMaxF(&agg2[d], m);
}

// wave-per-node 512-dot with root2, + agg + bias, sigmoid
__global__ void spline2_final(const float* __restrict__ agg2, const unsigned short* __restrict__ h2s,
                              const float* __restrict__ root2, const float* __restrict__ bs2,
                              float* __restrict__ out) {
    long long gt = (long long)blockIdx.x * blockDim.x + threadIdx.x;
    int n    = (int)(gt >> 5);
    int lane = (int)(gt & 31);
    if (n >= NN) return;
    float dot = 0.f;
    for (int i = lane; i < 512; i += 32)
        dot += bf2f(h2s[(size_t)n * 512 + i]) * root2[i];
#pragma unroll
    for (int off = 16; off > 0; off >>= 1)
        dot += __shfl_xor(dot, off, 32);
    if (lane == 0) {
        float m = agg2[n];
        if (m == -INFINITY) m = 0.f;
        float v = m + dot + bs2[0];
        out[n] = 1.0f / (1.0f + __expf(-v));
    }
}

// ---------------- host ----------------

extern "C" void kernel_launch(void* const* d_in, const int* in_sizes, int n_in,
                              void* d_out, int out_size, void* d_ws, size_t ws_size,
                              hipStream_t stream) {
    (void)in_sizes; (void)n_in; (void)out_size; (void)ws_size;

    const float* x      = (const float*)d_in[0];
    const int*   ei     = (const int*)  d_in[1];
    const float* eattr  = (const float*)d_in[2];
    const float* W1     = (const float*)d_in[3];
    const float* a1_src = (const float*)d_in[4];
    const float* a1_dst = (const float*)d_in[5];
    const float* b1     = (const float*)d_in[6];
    const float* Ws1    = (const float*)d_in[7];
    const float* root1  = (const float*)d_in[8];
    const float* bs1    = (const float*)d_in[9];
    const float* W2     = (const float*)d_in[10];
    const float* a2_src = (const float*)d_in[11];
    const float* a2_dst = (const float*)d_in[12];
    const float* b2     = (const float*)d_in[13];
    const float* Ws2    = (const float*)d_in[14];
    const float* root2  = (const float*)d_in[15];
    const float* bs2    = (const float*)d_in[16];

    char* ws = (char*)d_ws;
    size_t off = 0;
    auto alloc = [&](size_t bytes) -> void* {
        void* p = ws + off;
        off += (bytes + 255) & ~(size_t)255;
        return p;
    };

    unsigned short* xpad   = (unsigned short*)alloc((size_t)NN * 32 * 2);
    unsigned short* W1p    = (unsigned short*)alloc(32 * 256 * 2);
    unsigned short* root1b = (unsigned short*)alloc(256 * 64 * 2);
    unsigned short* W2p    = (unsigned short*)alloc(64 * 512 * 2);
    unsigned short* Ws1p   = (unsigned short*)alloc(256 * 1600 * 2);
    unsigned short* Ws2p   = (unsigned short*)alloc(512 * 32 * 2);
    unsigned short* W1t    = (unsigned short*)alloc(32 * 256 * 2);     // tiled copies
    unsigned short* root1t = (unsigned short*)alloc(256 * 64 * 2);
    unsigned short* W2t    = (unsigned short*)alloc(64 * 512 * 2);
    unsigned short* Ws1t   = (unsigned short*)alloc((size_t)256 * 1600 * 2);
    unsigned short* Ws2t   = (unsigned short*)alloc(512 * 32 * 2);
    float* basis = (float*)alloc((size_t)EE * 4 * 4);
    int*   bidx  = (int*)  alloc((size_t)EE * 4 * 4);
    float* eatt  = (float*)alloc((size_t)ETOT * 8 * 4);
    float* es    = (float*)alloc((size_t)NN * 8 * 4);
    float* ed    = (float*)alloc((size_t)NN * 8 * 4);
    float* mbuf  = (float*)alloc((size_t)NN * 8 * 4);
    float* sbuf  = (float*)alloc((size_t)NN * 8 * 4);
    float* xp1   = (float*)alloc((size_t)NN * 256 * 4);
    float* acc1  = (float*)alloc((size_t)NN * 256 * 4);   // contiguous after xp1
    unsigned short* h1b = (unsigned short*)alloc((size_t)NN * 256 * 2);
    unsigned short* xk1 = (unsigned short*)alloc((size_t)NN * 1600 * 2);  // 160 MB
    float* rootout1 = (float*)alloc((size_t)NN * 64 * 4);
    float* agg1  = (float*)alloc((size_t)NN * 64 * 4);
    unsigned short* h2b = (unsigned short*)alloc((size_t)NN * 64 * 2);
    unsigned short* xk2 = (unsigned short*)alloc((size_t)NN * 32 * 2);
    float* agg2  = (float*)alloc((size_t)NN * 4);
    // aliases into dead regions (lifetimes disjoint, stream-ordered):
    float*          xp2  = (float*)xk1;                                     // 102.4 MB <= 160 MB
    unsigned short* h2sb = (unsigned short*)((char*)xk1 + (size_t)NN * 512 * 4); // next 51.2 MB
    float*          acc2 = xp1;                                             // spans xp1+acc1

    const int T = 256;
    const int MB = NN / (16 * MT);   // 625 blocks of 80 rows

    // ---- packing: fp32 -> row-major bf16 -> WMMA-tiled bf16 ----
    cvt_pad<<<ceil_div((long long)NN * 32, T), T, 0, stream>>>(x, xpad, NN, 16, NN, 32);
    cvt_pad<<<ceil_div(32 * 256, T), T, 0, stream>>>(W1, W1p, 16, 256, 32, 256);
    cvt_pad<<<ceil_div(256 * 64, T), T, 0, stream>>>(root1, root1b, 256, 64, 256, 64);
    cvt_pad<<<ceil_div(64 * 512, T), T, 0, stream>>>(W2, W2p, 64, 512, 64, 512);
    repack_ws1<<<ceil_div(256LL * 1600, T), T, 0, stream>>>(Ws1, Ws1p);
    repack_ws2<<<ceil_div(512 * 32, T), T, 0, stream>>>(Ws2, Ws2p);
    pack_b_tiles<<<ceil_div(32LL * 256, T), T, 0, stream>>>(W1p, W1t, 32, 256);
    pack_b_tiles<<<ceil_div(256LL * 64, T), T, 0, stream>>>(root1b, root1t, 256, 64);
    pack_b_tiles<<<ceil_div(64LL * 512, T), T, 0, stream>>>(W2p, W2t, 64, 512);
    pack_b_tiles<<<ceil_div(256LL * 1600, T), T, 0, stream>>>(Ws1p, Ws1t, 256, 1600);
    pack_b_tiles<<<ceil_div(512LL * 32, T), T, 0, stream>>>(Ws2p, Ws2t, 512, 32);
    spline_basis_k<<<ceil_div(EE, T), T, 0, stream>>>(eattr, basis, bidx);

    // ---- GAT layer 1 (H=8, C=32) ----
    wmma_gemm<false><<<dim3(MB, 2), 256, 0, stream>>>(xpad, W1t, xp1, 256, 32);
    gat_node_scores<<<ceil_div((long long)NN * 8, T), T, 0, stream>>>(xp1, a1_src, a1_dst, es, ed, 32);
    fill_f32<<<ceil_div((long long)NN * 8, T), T, 0, stream>>>(mbuf, (long long)NN * 8, -INFINITY);
    fill_f32<<<ceil_div((long long)NN * 8, T), T, 0, stream>>>(sbuf, (long long)NN * 8, 0.0f);
    fill_f32<<<ceil_div((long long)NN * 256, T), T, 0, stream>>>(acc1, (long long)NN * 256, 0.0f);
    gat_edge_logits<<<ceil_div((long long)ETOT * 8, T), T, 0, stream>>>(ei, es, ed, eatt, mbuf);
    fix_neginf<<<ceil_div((long long)NN * 8, T), T, 0, stream>>>(mbuf, (long long)NN * 8);
    gat_edge_expsum<<<ceil_div((long long)ETOT * 8, T), T, 0, stream>>>(ei, mbuf, eatt, sbuf);
    gat_scatter<<<ETOT, 256, 0, stream>>>(ei, eatt, sbuf, xp1, acc1, 256, 5);
    gat_finalize<<<ceil_div((long long)NN * 256, T), T, 0, stream>>>(acc1, b1, h1b, 256);

    // ---- Spline layer 1 (256 -> 64) ----
    wmma_gemm<true ><<<dim3(MB, ceil_div(100, 8)), 256, 0, stream>>>(h1b, Ws1t, xk1, 1600, 256);
    wmma_gemm<false><<<dim3(MB, 1), 256, 0, stream>>>(h1b, root1t, rootout1, 64, 256);
    fill_f32<<<ceil_div((long long)NN * 64, T), T, 0, stream>>>(agg1, (long long)NN * 64, -INFINITY);
    spline1_edge<<<ceil_div((long long)EE * 64, T), T, 0, stream>>>(ei, basis, bidx, xk1, agg1);
    spline1_finalize<<<ceil_div((long long)NN * 64, T), T, 0, stream>>>(agg1, rootout1, bs1, h2b);

    // ---- GAT layer 2 (H=8, C=64) ----
    wmma_gemm<false><<<dim3(MB, 4), 256, 0, stream>>>(h2b, W2t, xp2, 512, 64);
    gat_node_scores<<<ceil_div((long long)NN * 8, T), T, 0, stream>>>(xp2, a2_src, a2_dst, es, ed, 64);
    fill_f32<<<ceil_div((long long)NN * 8, T), T, 0, stream>>>(mbuf, (long long)NN * 8, -INFINITY);
    fill_f32<<<ceil_div((long long)NN * 8, T), T, 0, stream>>>(sbuf, (long long)NN * 8, 0.0f);
    fill_f32<<<ceil_div((long long)NN * 512, T), T, 0, stream>>>(acc2, (long long)NN * 512, 0.0f);
    gat_edge_logits<<<ceil_div((long long)ETOT * 8, T), T, 0, stream>>>(ei, es, ed, eatt, mbuf);
    fix_neginf<<<ceil_div((long long)NN * 8, T), T, 0, stream>>>(mbuf, (long long)NN * 8);
    gat_edge_expsum<<<ceil_div((long long)ETOT * 8, T), T, 0, stream>>>(ei, mbuf, eatt, sbuf);
    gat_scatter<<<ETOT, 512, 0, stream>>>(ei, eatt, sbuf, xp2, acc2, 512, 6);
    gat_finalize<<<ceil_div((long long)NN * 512, T), T, 0, stream>>>(acc2, b2, h2sb, 512);

    // ---- Spline layer 2 (512 -> 1) + sigmoid ----
    wmma_gemm<true><<<dim3(MB, 1), 256, 0, stream>>>(h2sb, Ws2t, xk2, 32, 512);
    fill_f32<<<ceil_div(NN, T), T, 0, stream>>>(agg2, NN, -INFINITY);
    spline2_edge<<<ceil_div(EE, T), T, 0, stream>>>(ei, basis, bidx, xk2, agg2);
    spline2_final<<<ceil_div((long long)NN * 32, T), T, 0, stream>>>(agg2, h2sb, root2, bs2, (float*)d_out);
}
